// Mamba3D_35991825940956
// MI455X (gfx1250) — compile-verified, matrix operand order checked by reading, not run on hardware
//
#include <hip/hip_runtime.h>
#include <hip/hip_bf16.h>

// ---------------- types for CDNA5 WMMA ----------------
typedef __attribute__((ext_vector_type(16))) __bf16 v16bf;
typedef __attribute__((ext_vector_type(8)))  float  v8f;

// ---------------- model constants ----------------
#define S_TOK   2048      // L*H*W = 8*16*16
#define DM      384       // d_model
#define DI      768       // d_inner
#define DXP     64        // x_proj out padded (24 + 16 + 16 = 56 -> 64)
#define DTRANK  24
#define DTK     32        // dt_proj K padded
#define DST     16        // d_state
#define NBLK    6
#define NCH     32        // scan chunks
#define CL      64        // chunk length (S_TOK / NCH)

__device__ __forceinline__ unsigned short f32_to_bf16(float f) {
    unsigned int u = __float_as_uint(f);
    unsigned int r = u + 0x7FFFu + ((u >> 16) & 1u);   // round-to-nearest-even
    return (unsigned short)(r >> 16);
}
__device__ __forceinline__ __bf16 us_to_bf(unsigned short u) {
    union { unsigned short u; __bf16 b; } c; c.u = u; return c.b;
}
__device__ __forceinline__ float silu_f(float x) {
    return x / (1.f + __expf(-x));
}

// token id (canonical l*256 + h*16 + w) of sequence position s under ordering
// ord: 0 = hwl, 1 = lwh, 2 = lhw (identity); rev flips the sequence.
__device__ __forceinline__ int perm_token(int s, int ord, int rev) {
    if (rev) s = (S_TOK - 1) - s;
    int l, h, w;
    if (ord == 0) {          // s = h*128 + w*8 + l
        h = s >> 7; w = (s >> 3) & 15; l = s & 7;
    } else if (ord == 1) {   // s = l*256 + w*16 + h
        l = s >> 8; w = (s >> 4) & 15; h = s & 15;
    } else {                 // s = l*256 + h*16 + w
        l = s >> 8; h = (s >> 4) & 15; w = s & 15;
    }
    return l * 256 + h * 16 + w;
}

// ---------------- weight conversion ----------------
__global__ void cvt_bf16_kernel(const float* __restrict__ src,
                                unsigned short* __restrict__ dst, int n) {
    int i = blockIdx.x * blockDim.x + threadIdx.x;
    if (i < n) dst[i] = f32_to_bf16(src[i]);
}

// x_proj: (6,56,768) -> (6,64,768), rows 56..63 zero
__global__ void cvt_pad_xproj_kernel(const float* __restrict__ src,
                                     unsigned short* __restrict__ dst) {
    int i = blockIdx.x * blockDim.x + threadIdx.x;
    if (i >= NBLK * DXP * DI) return;
    int k = i % DI;
    int r = (i / DI) % DXP;
    int b = i / (DI * DXP);
    dst[i] = (r < 56) ? f32_to_bf16(src[((size_t)b * 56 + r) * DI + k])
                      : (unsigned short)0;
}

// dt_proj: (6,768,24) -> (6,768,32), cols 24..31 zero
__global__ void cvt_pad_dtproj_kernel(const float* __restrict__ src,
                                      unsigned short* __restrict__ dst) {
    int i = blockIdx.x * blockDim.x + threadIdx.x;
    if (i >= NBLK * DI * DTK) return;
    int k = i & 31;
    int rc = i >> 5;
    int r = rc % DI;
    int b = rc / DI;
    dst[i] = (k < DTRANK) ? f32_to_bf16(src[((size_t)b * DI + r) * DTRANK + k])
                          : (unsigned short)0;
}

__global__ void copy_f32_kernel(const float* __restrict__ src,
                                float* __restrict__ dst, int n) {
    int i = blockIdx.x * blockDim.x + threadIdx.x;
    if (i < n) dst[i] = src[i];
}

// ---------------- gather + RMSNorm -> bf16 ----------------
__global__ __launch_bounds__(128)
void gather_rmsnorm_kernel(const float* __restrict__ resid,
                           const float* __restrict__ nw,
                           unsigned short* __restrict__ hbf,
                           int ord, int rev) {
    const int s = blockIdx.x;
    const int t = perm_token(s, ord, rev);
    const float* xr = resid + (size_t)t * DM;
    __shared__ float red[128];
    float vals[3];
    float ss = 0.f;
#pragma unroll
    for (int i = 0; i < 3; ++i) {
        float v = xr[threadIdx.x + i * 128];
        vals[i] = v;
        ss += v * v;
    }
    red[threadIdx.x] = ss;
    __syncthreads();
    for (int off = 64; off > 0; off >>= 1) {
        if (threadIdx.x < off) red[threadIdx.x] += red[threadIdx.x + off];
        __syncthreads();
    }
    const float scale = rsqrtf(red[0] / (float)DM + 1e-5f);
#pragma unroll
    for (int i = 0; i < 3; ++i) {
        int d = threadIdx.x + i * 128;
        hbf[(size_t)s * DM + d] = f32_to_bf16(vals[i] * scale * nw[d]);
    }
}

// ---------------- bf16 WMMA GEMM: C[M,N] = A[M,K] x W[N,K]^T -----------------
// Each wave computes a 16x64 strip (4 N-tiles) so the A fragment is loaded once
// per K-step and reused across 4 WMMAs (register blocking; all data L2-resident).
// mode 0: store fp32
// mode 1: store softplus(val + bias[n])          (dt_proj)
// mode 2: resid[perm(m)*384 + n] += val          (out_proj residual scatter)
__global__ __launch_bounds__(256)
void wmma_gemm_bf16(const unsigned short* __restrict__ A,
                    const unsigned short* __restrict__ Bw,
                    float* __restrict__ Cout,
                    int M, int N, int K,
                    int mode, const float* __restrict__ bias,
                    float* __restrict__ resid, int ord, int rev) {
    const int wave = threadIdx.x >> 5;
    const int lane = threadIdx.x & 31;
    const int nstrips = N >> 6;                 // 64-wide output strips
    const int strips = (M >> 4) * nstrips;
    const int strip = blockIdx.x * 8 + wave;
    if (strip >= strips) return;
    const int ns = strip % nstrips;
    const int ms = strip / nstrips;

    // CDNA5 16-bit A-matrix layout: lanes 0-15 hold K 0-7 & 16-23,
    // lanes 16-31 hold K 8-15 & 24-31 (row = lane & 15).
    const int row   = lane & 15;
    const int khalf = lane >> 4;
    const unsigned short* ap  = A  + (size_t)(ms * 16 + row) * K + khalf * 8;
    const unsigned short* bp0 = Bw + (size_t)(ns * 64 + row) * K + khalf * 8;

    v8f acc[4];
#pragma unroll
    for (int t = 0; t < 4; ++t) acc[t] = (v8f){};

    for (int kk = 0; kk < K; kk += 32) {
        if (kk + 32 < K) {                      // gfx1250 global_prefetch_b8
            __builtin_prefetch(ap + kk + 32, 0, 1);
            __builtin_prefetch(bp0 + kk + 32, 0, 1);
        }
        v16bf a;
#pragma unroll
        for (int j = 0; j < 8; ++j) {
            a[j]     = us_to_bf(ap[kk + j]);
            a[j + 8] = us_to_bf(ap[kk + 16 + j]);
        }
#pragma unroll
        for (int t = 0; t < 4; ++t) {
            const unsigned short* bp = bp0 + (size_t)t * 16 * K;
            v16bf b;
#pragma unroll
            for (int j = 0; j < 8; ++j) {
                b[j]     = us_to_bf(bp[kk + j]);
                b[j + 8] = us_to_bf(bp[kk + 16 + j]);
            }
            acc[t] = __builtin_amdgcn_wmma_f32_16x16x32_bf16(
                         false, a, false, b, (short)0, acc[t], false, false);
        }
    }

    // C/D layout: lane n = col (lane&15); VGPR v -> row v + 8*(lane>>4)
    const int n  = lane & 15;
    const int m0 = (lane >> 4) * 8;
#pragma unroll
    for (int t = 0; t < 4; ++t) {
        const int gn = ns * 64 + t * 16 + n;
#pragma unroll
        for (int v = 0; v < 8; ++v) {
            const int gm = ms * 16 + m0 + v;
            float val = acc[t][v];
            if (mode == 0) {
                Cout[(size_t)gm * N + gn] = val;
            } else if (mode == 1) {
                float xx = val + bias[gn];
                Cout[(size_t)gm * N + gn] =
                    (xx > 20.f) ? xx : __logf(1.f + __expf(xx));
            } else {
                int tk = perm_token(gm, ord, rev);
                resid[(size_t)tk * DM + gn] += val;
            }
        }
    }
}

// ---------------- depthwise causal conv (k=4) + SiLU ----------------
__global__ void conv_silu_kernel(const float* __restrict__ xz,
                                 const float* __restrict__ cw,
                                 const float* __restrict__ cb,
                                 float* __restrict__ xp,
                                 unsigned short* __restrict__ xp_bf) {
    int idx = blockIdx.x * blockDim.x + threadIdx.x;
    if (idx >= S_TOK * DI) return;
    const int c = idx % DI;
    const int s = idx / DI;
    float acc = cb[c];
#pragma unroll
    for (int j = 0; j < 4; ++j) {
        int sp = s - 3 + j;
        float xv = (sp >= 0) ? xz[(size_t)sp * (2 * DI) + c] : 0.f;
        acc += cw[c * 4 + j] * xv;
    }
    float sv = silu_f(acc);
    xp[idx] = sv;
    xp_bf[idx] = f32_to_bf16(sv);
}

// ---------------- dt slice -> bf16 (K padded 24->32) ----------------
__global__ void dt_prep_kernel(const float* __restrict__ x_dbl,
                               unsigned short* __restrict__ dt_bf) {
    int idx = blockIdx.x * blockDim.x + threadIdx.x;
    if (idx >= S_TOK * DTK) return;
    int k = idx & 31;
    int s = idx >> 5;
    dt_bf[idx] = (k < DTRANK) ? f32_to_bf16(x_dbl[(size_t)s * DXP + k])
                              : (unsigned short)0;
}

// ---------------- chunked selective scan (3 passes) ------------------------
// h_s = a_s*h_{s-1} + b_s, elementwise over (c,j).
// Pass 1: per-chunk aggregates with h0=0: P = prod(a), H = local end-state.
// Pass 2: 32-step sequential combine over chunks -> initial state per chunk.
// Pass 3: replay chunk with true initial state, produce gated y (bf16).

__global__ __launch_bounds__(128)
void scan_pass1(const float* __restrict__ dtv,
                const float* __restrict__ xp,
                const float* __restrict__ x_dbl,
                const float* __restrict__ A_log,
                float* __restrict__ Pagg,
                float* __restrict__ Hagg) {
    const int q = blockIdx.x / (DI / 128);
    const int c = (blockIdx.x % (DI / 128)) * 128 + threadIdx.x;
    float A[DST], h[DST], P[DST];
#pragma unroll
    for (int j = 0; j < DST; ++j) {
        A[j] = -__expf(A_log[(size_t)c * DST + j]);
        h[j] = 0.f;
        P[j] = 1.f;
    }
    const int s0 = q * CL;
    for (int i = 0; i < CL; ++i) {
        const int s = s0 + i;
        const float dt = dtv[(size_t)s * DI + c];
        const float u  = xp [(size_t)s * DI + c];
        const float* bc = x_dbl + (size_t)s * DXP;
#pragma unroll
        for (int j = 0; j < DST; ++j) {
            float a = __expf(dt * A[j]);
            h[j] = a * h[j] + dt * bc[24 + j] * u;
            P[j] *= a;
        }
    }
    const size_t base = ((size_t)q * DI + c) * DST;
#pragma unroll
    for (int j = 0; j < DST; ++j) {
        Pagg[base + j] = P[j];
        Hagg[base + j] = h[j];
    }
}

__global__ __launch_bounds__(128)
void scan_pass2(const float* __restrict__ Pagg,
                const float* __restrict__ Hagg,
                float* __restrict__ Hinit) {
    const int t = blockIdx.x * 128 + threadIdx.x;    // < DI*DST = 12288
    float h = 0.f;
    for (int q = 0; q < NCH; ++q) {
        const size_t idx = (size_t)q * DI * DST + t;
        Hinit[idx] = h;
        h = Pagg[idx] * h + Hagg[idx];
    }
}

__global__ __launch_bounds__(128)
void scan_pass3(const float* __restrict__ dtv,
                const float* __restrict__ xp,
                const float* __restrict__ x_dbl,
                const float* __restrict__ xz,
                const float* __restrict__ A_log,
                const float* __restrict__ Dp,
                const float* __restrict__ Hinit,
                unsigned short* __restrict__ y_bf) {
    const int q = blockIdx.x / (DI / 128);
    const int c = (blockIdx.x % (DI / 128)) * 128 + threadIdx.x;
    float A[DST], h[DST];
    const size_t base = ((size_t)q * DI + c) * DST;
#pragma unroll
    for (int j = 0; j < DST; ++j) {
        A[j] = -__expf(A_log[(size_t)c * DST + j]);
        h[j] = Hinit[base + j];
    }
    const float D = Dp[c];
    const int s0 = q * CL;
    for (int i = 0; i < CL; ++i) {
        const int s = s0 + i;
        const float dt = dtv[(size_t)s * DI + c];
        const float u  = xp [(size_t)s * DI + c];
        const float* bc = x_dbl + (size_t)s * DXP;
        float acc = 0.f;
#pragma unroll
        for (int j = 0; j < DST; ++j) {
            h[j] = __expf(dt * A[j]) * h[j] + dt * bc[24 + j] * u;
            acc += h[j] * bc[40 + j];
        }
        float y = acc + u * D;
        const float z = xz[(size_t)s * (2 * DI) + DI + c];
        y *= silu_f(z);
        y_bf[(size_t)s * DI + c] = f32_to_bf16(y);
    }
}

// =======================================================================
extern "C" void kernel_launch(void* const* d_in, const int* in_sizes, int n_in,
                              void* d_out, int out_size, void* d_ws, size_t ws_size,
                              hipStream_t stream) {
    const float* x         = (const float*)d_in[0];
    const float* norm_w    = (const float*)d_in[1];
    const float* in_proj_w = (const float*)d_in[2];
    const float* conv_w    = (const float*)d_in[3];
    const float* conv_b    = (const float*)d_in[4];
    const float* x_proj_w  = (const float*)d_in[5];
    const float* dt_proj_w = (const float*)d_in[6];
    const float* dt_proj_b = (const float*)d_in[7];
    const float* A_log     = (const float*)d_in[8];
    const float* D_param   = (const float*)d_in[9];
    const float* out_proj_w= (const float*)d_in[10];
    float* out = (float*)d_out;

    char* ws = (char*)d_ws;
    size_t off = 0;
    auto alloc = [&](size_t bytes) -> char* {
        char* p = ws + off;
        off = (off + bytes + 255) & ~(size_t)255;
        return p;
    };

    float*          resid  = (float*)         alloc((size_t)S_TOK * DM * 4);
    unsigned short* hbf    = (unsigned short*)alloc((size_t)S_TOK * DM * 2);
    float*          xz     = (float*)         alloc((size_t)S_TOK * 2 * DI * 4);
    float*          xp     = (float*)         alloc((size_t)S_TOK * DI * 4);
    unsigned short* xp_bf  = (unsigned short*)alloc((size_t)S_TOK * DI * 2);
    float*          x_dbl  = (float*)         alloc((size_t)S_TOK * DXP * 4);
    unsigned short* dt_bf  = (unsigned short*)alloc((size_t)S_TOK * DTK * 2);
    float*          dtv    = (float*)         alloc((size_t)S_TOK * DI * 4);
    unsigned short* y_bf   = (unsigned short*)alloc((size_t)S_TOK * DI * 2);
    float*          Pagg   = (float*)         alloc((size_t)NCH * DI * DST * 4);
    float*          Hagg   = (float*)         alloc((size_t)NCH * DI * DST * 4);
    float*          Hinit  = (float*)         alloc((size_t)NCH * DI * DST * 4);
    unsigned short* w_in   = (unsigned short*)alloc((size_t)NBLK * 2 * DI * DM * 2);
    unsigned short* w_x    = (unsigned short*)alloc((size_t)NBLK * DXP * DI * 2);
    unsigned short* w_dt   = (unsigned short*)alloc((size_t)NBLK * DI * DTK * 2);
    unsigned short* w_out  = (unsigned short*)alloc((size_t)NBLK * DM * DI * 2);

    // --- weight conversion (fp32 -> bf16, with padding where needed) ---
    {
        int n = NBLK * 2 * DI * DM;
        cvt_bf16_kernel<<<(n + 255) / 256, 256, 0, stream>>>(in_proj_w, w_in, n);
        n = NBLK * DM * DI;
        cvt_bf16_kernel<<<(n + 255) / 256, 256, 0, stream>>>(out_proj_w, w_out, n);
        n = NBLK * DXP * DI;
        cvt_pad_xproj_kernel<<<(n + 255) / 256, 256, 0, stream>>>(x_proj_w, w_x);
        n = NBLK * DI * DTK;
        cvt_pad_dtproj_kernel<<<(n + 255) / 256, 256, 0, stream>>>(dt_proj_w, w_dt);
    }

    // --- residual stream init: canonical (l,h,w,D) layout == input layout ---
    copy_f32_kernel<<<(S_TOK * DM + 255) / 256, 256, 0, stream>>>(x, resid, S_TOK * DM);

    const int scan_blocks = NCH * (DI / 128);   // 192

    for (int b = 0; b < NBLK; ++b) {
        const int ord = b >> 1;     // 0,0,1,1,2,2
        const int rev = b & 1;      // fwd, rev, fwd, rev, ...

        // gather (permute) + RMSNorm -> bf16
        gather_rmsnorm_kernel<<<S_TOK, 128, 0, stream>>>(
            resid, norm_w + (size_t)b * DM, hbf, ord, rev);

        // in_proj: (2048x384) x (1536x384)^T -> xz (2048x1536)
        {
            int strips = (S_TOK / 16) * (2 * DI / 64);
            wmma_gemm_bf16<<<(strips + 7) / 8, 256, 0, stream>>>(
                hbf, w_in + (size_t)b * 2 * DI * DM, xz,
                S_TOK, 2 * DI, DM, 0, nullptr, nullptr, 0, 0);
        }

        // depthwise causal conv(k=4) + SiLU on first half of xz
        conv_silu_kernel<<<(S_TOK * DI + 255) / 256, 256, 0, stream>>>(
            xz, conv_w + (size_t)b * DI * 4, conv_b + (size_t)b * DI, xp, xp_bf);

        // x_proj: (2048x768) x (64x768)^T -> x_dbl (2048x64)  [dt|B|C padded]
        {
            int strips = (S_TOK / 16) * (DXP / 64);
            wmma_gemm_bf16<<<(strips + 7) / 8, 256, 0, stream>>>(
                xp_bf, w_x + (size_t)b * DXP * DI, x_dbl,
                S_TOK, DXP, DI, 0, nullptr, nullptr, 0, 0);
        }

        // dt slice -> bf16 (K pad 24->32)
        dt_prep_kernel<<<(S_TOK * DTK + 255) / 256, 256, 0, stream>>>(x_dbl, dt_bf);

        // dt_proj + bias + softplus: (2048x32) x (768x32)^T -> dtv (2048x768)
        {
            int strips = (S_TOK / 16) * (DI / 64);
            wmma_gemm_bf16<<<(strips + 7) / 8, 256, 0, stream>>>(
                dt_bf, w_dt + (size_t)b * DI * DTK, dtv,
                S_TOK, DI, DTK, 1, dt_proj_b + (size_t)b * DI, nullptr, 0, 0);
        }

        // chunked selective scan + SiLU(z) gate -> y bf16
        scan_pass1<<<scan_blocks, 128, 0, stream>>>(
            dtv, xp, x_dbl, A_log + (size_t)b * DI * DST, Pagg, Hagg);
        scan_pass2<<<(DI * DST) / 128, 128, 0, stream>>>(Pagg, Hagg, Hinit);
        scan_pass3<<<scan_blocks, 128, 0, stream>>>(
            dtv, xp, x_dbl, xz,
            A_log + (size_t)b * DI * DST, D_param + (size_t)b * DI, Hinit, y_bf);

        // out_proj + permuted residual scatter-add
        {
            int strips = (S_TOK / 16) * (DM / 64);
            wmma_gemm_bf16<<<(strips + 7) / 8, 256, 0, stream>>>(
                y_bf, w_out + (size_t)b * DM * DI, nullptr,
                S_TOK, DM, DI, 2, nullptr, resid, ord, rev);
        }
    }

    // final output: residual stream already in (N,L,H,W,D) canonical order
    copy_f32_kernel<<<(S_TOK * DM + 255) / 256, 256, 0, stream>>>(resid, out, S_TOK * DM);
}
